// SSDIR_9431748182652
// MI455X (gfx1250) — matrix operand order repaired and checked by reading, not compiled, not run on hardware
//
#include <hip/hip_runtime.h>
#include <hip/hip_bf16.h>
#include <cstdint>

// ---------------------------------------------------------------------------
// SSDIR decoder for MI455X (gfx1250, wave32, WMMA).
//  * conv-transpose(2x2, stride2, VALID) == GEMM  X[M x Cin] * W[Cin x 4*Cout]
//  * all six layers run through v_wmma_f32_16x16x32_f16 (f16 in, f32 acc)
//  * decoded patches kept in f16 (37 MB -> L2 resident on 192 MB L2)
//  * STN + softmax blend fused into one gather pass (never materialize recon)
// ---------------------------------------------------------------------------

typedef __attribute__((ext_vector_type(16))) _Float16 v16h;
typedef __attribute__((ext_vector_type(8)))  _Float16 v8h;
typedef __attribute__((ext_vector_type(8)))  float    v8f;

#define N_BOX 380
#define N_LOC 85
#define BATCH 4
#define ZW    64

__device__ __forceinline__ int locidx(int n) {
  // recon indices: fm8 x4 bpl, fm4 x6, fm2 x6, fm1 x4
  if (n < 256) return n >> 2;
  if (n < 352) return 64 + (n - 256) / 6;
  if (n < 376) return 80 + (n - 352) / 6;
  return 84;
}

// ---- weights f32 -> f16, pre-swizzled into exact WMMA B-fragment order ----
// dst[((ntile*KCH + kc)*32 + lane)*16 + e] = W[k][n] ; k = kc*32 + (lane<16?0:8)+(e<8?e:e+8)
// n = ntile*16 + (lane&15).  Zero-padded in K (layer 5: 16->32) and N (12->16).
__global__ void prep_w_kernel(const float* __restrict__ w, _Float16* __restrict__ dst,
                              int K, int COUT, int N4REAL, int KCHUNKS, int total) {
  int idx = blockIdx.x * blockDim.x + threadIdx.x;
  if (idx >= total) return;
  int e    = idx & 15;
  int lane = (idx >> 4) & 31;
  int kc   = (idx >> 9) % KCHUNKS;
  int nt   = idx / (512 * KCHUNKS);
  int n = nt * 16 + (lane & 15);
  int k = kc * 32 + ((lane >> 4) << 3) + (e < 8 ? e : e + 8);
  float v = 0.f;
  if (k < K && n < N4REAL) {
    int sub  = n / COUT;            // di*2+dj
    int cout = n % COUT;
    v = w[((size_t)(sub * K + k)) * COUT + cout];   // (2,2,Cin,Cout) HWIO
  }
  dst[idx] = (_Float16)v;
}

// ---- gather z_what per box, convert to f16: act0[(b*380+n)*64 + k] ----
__global__ void gather_z_kernel(const float* __restrict__ z_what,
                                _Float16* __restrict__ act0) {
  int idx = blockIdx.x * blockDim.x + threadIdx.x;
  if (idx >= BATCH * N_BOX * ZW) return;
  int k = idx & 63;
  int n = (idx >> 6) % N_BOX;
  int b = idx / (N_BOX * ZW);
  act0[idx] = (_Float16)z_what[((size_t)b * N_LOC + locidx(n)) * ZW + k];
}

// ---- WMMA GEMM layer: D = X * W, fused bias + (ReLU|sigmoid) + 2x-upsample
//      scatter-store of the next layer's activations (channels-last f16). ----
template<int K, int KPAD, int COUT, int N4REAL, int PIN, int WIN, bool SIG>
__global__ __launch_bounds__(256) void gemm_layer(
    const _Float16* __restrict__ actIn,   // [M][K] rows = img*PIN + pos
    const _Float16* __restrict__ wfrag,   // B-fragment-major, [Ntiles][KCH][32][16]
    const float*    __restrict__ bias,    // [COUT]
    _Float16*       __restrict__ actOut,  // [img][4*PIN][COUT]
    int Mtiles)
{
  constexpr int KCH = KPAD / 32;
  const int lane  = threadIdx.x & 31;
  const int wave  = threadIdx.x >> 5;
  const int mtile = blockIdx.x * 8 + wave;   // wave-uniform -> EXEC all-1 at WMMA
  const int ntile = blockIdx.y;
  if (mtile >= Mtiles) return;

  const int rowA = mtile * 16 + (lane & 15);
  const int aoff = (lane >> 4) << 3;                 // ISA A layout: 0 or 8
  const _Float16* aBase = actIn + (size_t)rowA * K + aoff;
  const _Float16* bBase = wfrag + (((size_t)ntile * KCH) * 32 + lane) * 16;

  v8f c;
  #pragma unroll
  for (int i = 0; i < 8; ++i) c[i] = 0.f;

  #pragma unroll
  for (int kc = 0; kc < KCH; ++kc) {
    // A: lane<16 holds K {kk..kk+7, kk+16..kk+23}; lane>=16 shifted by 8
    v8h lo = *(const v8h*)(aBase + kc * 32);
    v8h hi;
    if constexpr (K == KPAD) {
      hi = *(const v8h*)(aBase + kc * 32 + 16);
    } else {                                   // layer 5: K=16 zero-padded
      #pragma unroll
      for (int i = 0; i < 8; ++i) hi[i] = (_Float16)0.f;
    }
    v16h a;
    #pragma unroll
    for (int i = 0; i < 8; ++i) { a[i] = lo[i]; a[i + 8] = hi[i]; }
    v16h b = *(const v16h*)(bBase + (size_t)kc * 512);  // one 32B load / lane
    c = __builtin_amdgcn_wmma_f32_16x16x32_f16(false, a, false, b,
                                               (short)0, c, false, false);
  }

  const int col = ntile * 16 + (lane & 15);
  if (col < N4REAL) {
    const int sub  = col / COUT;
    const int cout = col % COUT;
    const int di = sub >> 1, dj = sub & 1;
    const float bv = bias[cout];
    #pragma unroll
    for (int r = 0; r < 8; ++r) {
      int row = mtile * 16 + r + ((lane >> 4) << 3);   // C/D layout
      int img = row / PIN;
      int p   = row - img * PIN;
      int i   = p / WIN;
      int j   = p - i * WIN;
      int opos = (2 * i + di) * (2 * WIN) + (2 * j + dj);
      float v = c[r] + bv;
      if constexpr (SIG) v = 1.f / (1.f + __expf(-v));
      else               v = fmaxf(v, 0.f);
      actOut[((size_t)img * (4 * PIN) + opos) * COUT + cout] = (_Float16)v;
    }
  }
}

// ---- softmax over box depths + per-box affine sampling params ----
// params[(b*380+n)*8] = {weight, Ax, Bx, Ay, By}
__global__ __launch_bounds__(512) void box_params_kernel(
    const float* __restrict__ z_where, const int* __restrict__ z_present,
    const float* __restrict__ z_depth, float* __restrict__ params)
{
  __shared__ float red[512];
  const int b = blockIdx.x;
  const int t = threadIdx.x;
  float d = -1e30f;
  if (t < N_BOX) {
    d = (z_present[b * N_BOX + t] == 1) ? z_depth[b * N_LOC + locidx(t)] : -1000.f;
  }
  red[t] = d;
  __syncthreads();
  for (int s = 256; s > 0; s >>= 1) {
    if (t < s) red[t] = fmaxf(red[t], red[t + s]);
    __syncthreads();
  }
  float mx = red[0];
  __syncthreads();
  float e = (t < N_BOX) ? __expf(d - mx) : 0.f;
  red[t] = e;
  __syncthreads();
  for (int s = 256; s > 0; s >>= 1) {
    if (t < s) red[t] += red[t + s];
    __syncthreads();
  }
  float inv = 1.f / red[0];
  if (t < N_BOX) {
    float wt = e * inv;
    const float* zw = z_where + ((size_t)(b * N_BOX + t)) * 4;
    float cx = zw[0], cy = zw[1], w = zw[2], h = zw[3];
    float gx = 31.5f / fmaxf(w, 1e-5f);
    float gy = 31.5f / fmaxf(h, 1e-5f);
    float* P = params + ((size_t)(b * N_BOX + t)) * 8;
    P[0] = wt;
    P[1] = gx * (2.f / 127.f);        // px = Ax*x + Bx
    P[2] = 31.5f - gx * 2.f * cx;
    P[3] = gy * (2.f / 127.f);        // py = Ay*y + By
    P[4] = 31.5f - gy * 2.f * cy;
  }
}

// ---- fused STN + weighted blend; decoded (f16, 37MB) stays L2-resident ----
__global__ __launch_bounds__(128) void stn_blend_kernel(
    const _Float16* __restrict__ decoded,  // [b*380+n][64][64][3] f16
    const float*    __restrict__ params,
    float*          __restrict__ out)      // [4][3][128][128] f32
{
  const int y = blockIdx.x;
  const int b = blockIdx.y;
  const int x = threadIdx.x;
  float acc0 = 0.f, acc1 = 0.f, acc2 = 0.f;
  for (int n = 0; n < N_BOX; ++n) {
    const float* P = params + ((size_t)(b * N_BOX + n)) * 8;
    float wt = P[0];
    if (!(wt > 0.f)) continue;                          // exact underflow skip
    float py = P[3] * (float)y + P[4];
    if (py <= -1.f || py >= 64.f) continue;             // block-uniform row skip
    float px = P[1] * (float)x + P[2];
    float x0f = floorf(px), y0f = floorf(py);
    float wx = px - x0f, wy = py - y0f;
    int x0 = (int)x0f, y0 = (int)y0f;
    const _Float16* img = decoded + (size_t)(b * N_BOX + n) * (4096 * 3);
    float w00 = wt * (1.f - wy) * (1.f - wx);
    float w01 = wt * (1.f - wy) * wx;
    float w10 = wt * wy * (1.f - wx);
    float w11 = wt * wy * wx;
    #pragma unroll
    for (int ty = 0; ty < 2; ++ty) {
      int yy = y0 + ty;
      if (yy < 0 || yy > 63) continue;
      #pragma unroll
      for (int tx = 0; tx < 2; ++tx) {
        int xx = x0 + tx;
        if (xx < 0 || xx > 63) continue;
        float tw = ty ? (tx ? w11 : w10) : (tx ? w01 : w00);
        const _Float16* q = img + (yy * 64 + xx) * 3;
        acc0 += tw * (float)q[0];
        acc1 += tw * (float)q[1];
        acc2 += tw * (float)q[2];
      }
    }
  }
  size_t ob = ((size_t)b * 3) * 16384 + (size_t)y * 128 + x;
  out[ob]               = acc0;
  out[ob + 16384]       = acc1;
  out[ob + 2 * 16384]   = acc2;
}

extern "C" void kernel_launch(void* const* d_in, const int* in_sizes, int n_in,
                              void* d_out, int out_size, void* d_ws, size_t ws_size,
                              hipStream_t stream)
{
  const float* z_what    = (const float*)d_in[0];
  const float* z_where   = (const float*)d_in[1];
  const int*   z_present = (const int*)  d_in[2];
  const float* z_depth   = (const float*)d_in[3];
  const float* W[6]; const float* Bv[6];
  for (int i = 0; i < 6; ++i) { W[i] = (const float*)d_in[4 + 2 * i];
                                Bv[i] = (const float*)d_in[5 + 2 * i]; }

  // ---- workspace layout (~88 MB total) ----
  char* ws = (char*)d_ws;
  size_t off = 0;
  _Float16* wf[6];
  const int wlen[6] = {65536, 131072, 32768, 8192, 2048, 512};  // KPAD*N4PAD halves
  for (int i = 0; i < 6; ++i) { wf[i] = (_Float16*)(ws + off); off += (size_t)wlen[i] * 2; }
  off = (off + 255) & ~(size_t)255;
  float* params = (float*)(ws + off); off += (size_t)BATCH * N_BOX * 8 * 4;
  off = (off + 255) & ~(size_t)255;
  _Float16* actA = (_Float16*)(ws + off); off += (size_t)18677760 * 2;  // max 37.4 MB
  off = (off + 255) & ~(size_t)255;
  _Float16* actB = (_Float16*)(ws + off);                               // max 49.8 MB

  // ---- weight swizzle + latent gather (cheap, every call: ws not persistent) ----
  const int Kl[6]  = {64, 256, 128, 64, 32, 16};
  const int Cl[6]  = {256, 128, 64, 32, 16, 3};
  const int N4r[6] = {1024, 512, 256, 128, 64, 12};
  const int KCH[6] = {2, 8, 4, 2, 1, 1};
  for (int l = 0; l < 6; ++l)
    prep_w_kernel<<<(wlen[l] + 255) / 256, 256, 0, stream>>>(
        W[l], wf[l], Kl[l], Cl[l], N4r[l], KCH[l], wlen[l]);
  gather_z_kernel<<<(BATCH * N_BOX * ZW + 255) / 256, 256, 0, stream>>>(z_what, actA);

  // ---- decoder: 6 WMMA GEMM layers (ping-pong actA/actB) ----
  //                 K  KPAD COUT N4R  PIN WIN SIG      grid(Mtiles/8, Ntiles)
  gemm_layer< 64,  64, 256, 1024,   1,  1, false><<<dim3(   12, 64), 256, 0, stream>>>(actA, wf[0], Bv[0], actB,    95);
  gemm_layer<256, 256, 128,  512,   4,  2, false><<<dim3(   48, 32), 256, 0, stream>>>(actB, wf[1], Bv[1], actA,   380);
  gemm_layer<128, 128,  64,  256,  16,  4, false><<<dim3(  190, 16), 256, 0, stream>>>(actA, wf[2], Bv[2], actB,  1520);
  gemm_layer< 64,  64,  32,  128,  64,  8, false><<<dim3(  760,  8), 256, 0, stream>>>(actB, wf[3], Bv[3], actA,  6080);
  gemm_layer< 32,  32,  16,   64, 256, 16, false><<<dim3( 3040,  4), 256, 0, stream>>>(actA, wf[4], Bv[4], actB, 24320);
  gemm_layer< 16,  32,   3,   12,1024, 32, true ><<<dim3(12160,  1), 256, 0, stream>>>(actB, wf[5], Bv[5], actA, 97280);

  // ---- softmax weights + affine params, then fused STN blend ----
  box_params_kernel<<<BATCH, 512, 0, stream>>>(z_where, z_present, z_depth, params);
  stn_blend_kernel<<<dim3(128, BATCH), 128, 0, stream>>>(actA, params, (float*)d_out);
}
